// GAT_24498493456645
// MI455X (gfx1250) — matrix operand order repaired and checked by reading, not compiled, not run on hardware
//
#include <hip/hip_runtime.h>
#include <hip/hip_bf16.h>

// ---------------------------------------------------------------------------
// GAT forward for MI455X (gfx1250, wave32, WMMA).
// All heavy math (x@W and att@h) runs on v_wmma_f32_16x16x32_f16.
// Attention probabilities are generated directly in the WMMA A-operand
// register layout (flash-style streaming softmax), so the [B,N,N] attention
// matrix is never materialized.
// ---------------------------------------------------------------------------

typedef __attribute__((ext_vector_type(16))) _Float16 v16h;
typedef __attribute__((ext_vector_type(8)))  float    v8f;

static constexpr int Bn    = 8;     // batch
static constexpr int Nn    = 1024;  // nodes
static constexpr int F0    = 256;   // nfeat
static constexpr int NH    = 64;    // nhid
static constexpr int HEADS = 8;
static constexpr int HID   = 512;   // NH*HEADS
static constexpr int NOUT  = 128;
static constexpr float ALPHA = 0.2f;

// A-operand K index for element i (i = 2v+s), per ISA 16-bit A 16x32 layout.
__device__ __forceinline__ int kmapA(int i, int hlf) {
    int v = i >> 1, s = i & 1;
    int base = (v < 4) ? (2 * v + s) : (16 + 2 * (v - 4) + s);
    return base + 8 * hlf;
}

// ---------------------------------------------------------------------------
// Adjacency bitmask: adj[b][i][w] bit j-in-word set if dist(i,j) < 1.0
// ---------------------------------------------------------------------------
__global__ void adj_kernel(const float* __restrict__ coord, unsigned* __restrict__ adj) {
    int t = blockIdx.x * blockDim.x + threadIdx.x;     // Bn*Nn*32 threads
    int w = t & 31;
    int i = (t >> 5) & (Nn - 1);
    int b = t >> 15;
    const float* ci = coord + ((size_t)b * Nn + i) * 3;
    float xi = ci[0], yi = ci[1], zi = ci[2];
    unsigned m = 0u;
    #pragma unroll 4
    for (int jj = 0; jj < 32; ++jj) {
        int j = w * 32 + jj;
        const float* cj = coord + ((size_t)b * Nn + j) * 3;
        float dx = xi - cj[0], dy = yi - cj[1], dz = zi - cj[2];
        float d2 = dx * dx + dy * dy + dz * dz;
        m |= (d2 < 1.0f ? 1u : 0u) << jj;
    }
    adj[t] = m;
}

// ---------------------------------------------------------------------------
// Column mean over nodes: ybar[b][f] = mean_n X[b][n][f]
// ---------------------------------------------------------------------------
__global__ void colmean_kernel(const float* __restrict__ X, float* __restrict__ ybar, int F) {
    int b = blockIdx.x, f = threadIdx.x;
    float s = 0.0f;
    for (int n = 0; n < Nn; ++n) s += X[((size_t)b * Nn + n) * F + f];
    ybar[b * F + f] = s * (1.0f / (float)Nn);
}

// ---------------------------------------------------------------------------
// SE MLP: scale[b][f] = sigmoid( relu(ybar @ w1^T) @ w2^T )
// w1: [Fr][F], w2: [F][Fr]; one block per batch, blockDim = F.
// ---------------------------------------------------------------------------
__global__ void se_mlp_kernel(const float* __restrict__ ybar, const float* __restrict__ w1,
                              const float* __restrict__ w2, float* __restrict__ scale,
                              int F, int Fr) {
    extern __shared__ float se_hid[];
    int b = blockIdx.x, t = threadIdx.x;
    if (t < Fr) {
        float s = 0.0f;
        for (int f = 0; f < F; ++f) s += ybar[b * F + f] * w1[t * F + f];
        se_hid[t] = s > 0.0f ? s : 0.0f;
    }
    __syncthreads();
    float s = 0.0f;
    for (int k = 0; k < Fr; ++k) s += se_hid[k] * w2[t * Fr + k];
    scale[b * F + t] = 1.0f / (1.0f + __expf(-s));
}

// ---------------------------------------------------------------------------
// Apply per-(b,f) SE scale and cast to f16 (WMMA A storage)
// ---------------------------------------------------------------------------
__global__ void scale_h16_kernel(const float* __restrict__ X, const float* __restrict__ scale,
                                 _Float16* __restrict__ out, int F, int total) {
    int idx = blockIdx.x * blockDim.x + threadIdx.x;
    if (idx >= total) return;
    int f = idx % F;
    int b = idx / (Nn * F);
    out[idx] = (_Float16)(X[idx] * scale[b * F + f]);
}

// ---------------------------------------------------------------------------
// WMMA GEMM: Hout(f16 [M][Fout]) = A(f16 [M][K]) @ W(f32 [K][Fout])
// One 16x16 output tile per wave; 4 waves / block; grid covers tiles exactly.
// ---------------------------------------------------------------------------
__global__ void gemm_h16_kernel(const _Float16* __restrict__ A, const float* __restrict__ W,
                                _Float16* __restrict__ Hout, int M, int K, int Fout) {
    int lane = threadIdx.x & 31;
    int wave = threadIdx.x >> 5;
    int tile = blockIdx.x * 4 + wave;
    int tilesN = Fout >> 4;
    int m0 = (tile / tilesN) * 16;
    int n0 = (tile % tilesN) * 16;
    int hlf = lane >> 4, mrow = lane & 15, ncol = mrow;

    v8f acc = {};
    for (int k0 = 0; k0 < K; k0 += 32) {
        v16h a, bv;
        #pragma unroll
        for (int i = 0; i < 16; ++i) {
            a[i]  = A[(size_t)(m0 + mrow) * K + (k0 + kmapA(i, hlf))];
            bv[i] = (_Float16)W[(size_t)(k0 + kmapA(i, hlf)) * Fout + n0 + ncol];
        }
        acc = __builtin_amdgcn_wmma_f32_16x16x32_f16(false, a, false, bv,
                                                     (short)0, acc, false, false);
    }
    #pragma unroll
    for (int j = 0; j < 8; ++j) {
        int row = m0 + j + 8 * hlf;
        Hout[(size_t)row * Fout + n0 + ncol] = (_Float16)acc[j];
    }
}

// ---------------------------------------------------------------------------
// e_src[m] = h[m]·a[:Fout],  e_dst[m] = h[m]·a[Fout:]
// ---------------------------------------------------------------------------
__global__ void edot_kernel(const _Float16* __restrict__ Hh, const float* __restrict__ av,
                            float* __restrict__ eS, float* __restrict__ eD, int M, int Fout) {
    int m = blockIdx.x * blockDim.x + threadIdx.x;
    if (m >= M) return;
    float s = 0.0f, d = 0.0f;
    for (int k = 0; k < Fout; ++k) {
        float h = (float)Hh[(size_t)m * Fout + k];
        s += h * av[k];
        d += h * av[Fout + k];
    }
    eS[m] = s; eD[m] = d;
}

// ---------------------------------------------------------------------------
// Attention: out_rows = elu( softmax(mask(leaky(e_src+e_dst))) @ h )
// One 16-row tile per wave; probs generated directly in WMMA A layout.
// Hh/eS/eD are per-head contiguous; out gets head*FOUT column offset.
// ---------------------------------------------------------------------------
template <int FOUT>
__global__ void attn_kernel(const _Float16* __restrict__ Hh, const float* __restrict__ eS,
                            const float* __restrict__ eD, const unsigned* __restrict__ adj,
                            float* __restrict__ out, int ostride) {
    const size_t M = (size_t)Bn * Nn;
    int lane = threadIdx.x & 31;
    int wave = threadIdx.x >> 5;
    int tile = blockIdx.x * 4 + wave;           // head*Bn*(Nn/16) tiles total
    int head = tile / (Bn * (Nn / 16));
    int rem  = tile % (Bn * (Nn / 16));
    int b    = rem / (Nn / 16);
    int i0   = (rem % (Nn / 16)) * 16;
    int hlf  = lane >> 4, mrow = lane & 15, ncol = mrow;

    const _Float16* Hp  = Hh + (size_t)head * M * FOUT;
    const float*    eSp = eS + (size_t)head * M;
    const float*    eDp = eD + (size_t)head * M;
    const unsigned* arow = adj + ((size_t)b * Nn + i0 + mrow) * (Nn / 32);
    const float es = eSp[(size_t)b * Nn + i0 + mrow];

    // Pass 1: masked row max (lane scans its half of the columns of its row)
    float mx = -3.0e38f;
    int jbase = hlf * (Nn / 2);
    for (int jj = 0; jj < Nn / 2; ++jj) {
        int j = jbase + jj;
        float e = es + eDp[(size_t)b * Nn + j];
        e = e > 0.0f ? e : ALPHA * e;
        bool on = (arow[j >> 5] >> (j & 31)) & 1u;
        mx = fmaxf(mx, on ? e : -3.0e38f);
    }
    mx = fmaxf(mx, __shfl_xor(mx, 16, 32));

    // Pass 2: stream K tiles of 32 columns through WMMA
    v8f acc[FOUT / 16] = {};
    float rsum = 0.0f;
    for (int j0 = 0; j0 < Nn; j0 += 32) {
        v16h a;
        #pragma unroll
        for (int i = 0; i < 16; ++i) {
            int j = j0 + kmapA(i, hlf);
            float e = es + eDp[(size_t)b * Nn + j];
            e = e > 0.0f ? e : ALPHA * e;
            bool on = (arow[j >> 5] >> (j & 31)) & 1u;
            float p = on ? __expf(e - mx) : 0.0f;
            rsum += p;
            a[i] = (_Float16)p;
        }
        #pragma unroll
        for (int c = 0; c < FOUT / 16; ++c) {
            v16h bv;
            #pragma unroll
            for (int i = 0; i < 16; ++i) {
                int j = j0 + kmapA(i, hlf);
                bv[i] = Hp[((size_t)b * Nn + j) * FOUT + c * 16 + ncol];
            }
            acc[c] = __builtin_amdgcn_wmma_f32_16x16x32_f16(false, a, false, bv,
                                                            (short)0, acc[c], false, false);
        }
    }
    rsum += __shfl_xor(rsum, 16, 32);           // lane L holds total for row L&15

    #pragma unroll
    for (int j = 0; j < 8; ++j) {
        int row = j + 8 * hlf;                  // C/D layout row for VGPR j
        float denom = __shfl(rsum, row, 32);
        #pragma unroll
        for (int c = 0; c < FOUT / 16; ++c) {
            float v = acc[c][j] / denom;
            v = v > 0.0f ? v : (__expf(v) - 1.0f);   // ELU
            out[((size_t)b * Nn + i0 + row) * ostride + head * FOUT + c * 16 + ncol] = v;
        }
    }
}

// ---------------------------------------------------------------------------
extern "C" void kernel_launch(void* const* d_in, const int* in_sizes, int n_in,
                              void* d_out, int out_size, void* d_ws, size_t ws_size,
                              hipStream_t stream) {
    (void)in_sizes; (void)n_in; (void)out_size; (void)ws_size;
    const float* x      = (const float*)d_in[0];   // [8,1024,256]
    const float* coord  = (const float*)d_in[1];   // [8,1024,3]
    const float* se1w1  = (const float*)d_in[2];   // [64,256]
    const float* se1w2  = (const float*)d_in[3];   // [256,64]
    const float* Ws     = (const float*)d_in[4];   // [8,256,64]
    const float* As     = (const float*)d_in[5];   // [8,128]
    const float* se2w1  = (const float*)d_in[6];   // [128,512]
    const float* se2w2  = (const float*)d_in[7];   // [512,128]
    const float* Wout   = (const float*)d_in[8];   // [512,128]
    const float* Aout   = (const float*)d_in[9];   // [256]

    const size_t M = (size_t)Bn * Nn;              // 8192 rows

    char* wp = (char*)d_ws;
    auto carve = [&](size_t bytes) {
        char* p = wp;
        wp += (bytes + 255) & ~(size_t)255;
        return (void*)p;
    };
    unsigned*  adj    = (unsigned*)carve((size_t)Bn * Nn * (Nn / 32) * 4); // 1 MiB
    float*     ybar1  = (float*)carve((size_t)Bn * F0 * 4);
    float*     scale1 = (float*)carve((size_t)Bn * F0 * 4);
    _Float16*  x1h    = (_Float16*)carve(M * F0 * 2);                      // 4 MiB
    _Float16*  h16    = (_Float16*)carve((size_t)HEADS * M * NH * 2);      // 8 MiB
    float*     eS1    = (float*)carve((size_t)HEADS * M * 4);
    float*     eD1    = (float*)carve((size_t)HEADS * M * 4);
    float*     x2     = (float*)carve(M * HID * 4);                        // 16 MiB
    float*     ybar2  = (float*)carve((size_t)Bn * HID * 4);
    float*     scale2 = (float*)carve((size_t)Bn * HID * 4);
    _Float16*  x2h    = (_Float16*)carve(M * HID * 2);                     // 8 MiB
    _Float16*  hout   = (_Float16*)carve(M * NOUT * 2);                    // 2 MiB
    float*     eS2    = (float*)carve(M * 4);
    float*     eD2    = (float*)carve(M * 4);

    // 1. adjacency bitmask
    adj_kernel<<<(Bn * Nn * 32) / 256, 256, 0, stream>>>(coord, adj);

    // 2. SE layer 1 on x [.,.,256]
    colmean_kernel<<<Bn, F0, 0, stream>>>(x, ybar1, F0);
    se_mlp_kernel<<<Bn, F0, (F0 / 4) * sizeof(float), stream>>>(ybar1, se1w1, se1w2, scale1, F0, F0 / 4);
    {
        int total = (int)(M * F0);
        scale_h16_kernel<<<(total + 255) / 256, 256, 0, stream>>>(x, scale1, x1h, F0, total);
    }

    // 3. per-head h = x1 @ W_h  (f16 WMMA GEMM), then e_src/e_dst dots
    for (int h = 0; h < HEADS; ++h) {
        const float* Wh = Ws + (size_t)h * F0 * NH;
        _Float16* Hh = h16 + (size_t)h * M * NH;
        int tiles = (int)(M / 16) * (NH / 16);               // 2048
        gemm_h16_kernel<<<tiles / 4, 128, 0, stream>>>(x1h, Wh, Hh, (int)M, F0, NH);
        edot_kernel<<<(int)(M / 256), 256, 0, stream>>>(Hh, As + (size_t)h * 2 * NH,
                                                        eS1 + (size_t)h * M, eD1 + (size_t)h * M,
                                                        (int)M, NH);
    }

    // 4. attention for all 8 heads, ELU fused, concat into x2 [.,.,512]
    {
        int tiles = HEADS * Bn * (Nn / 16);                  // 4096
        attn_kernel<NH><<<tiles / 4, 128, 0, stream>>>(h16, eS1, eD1, adj, x2, HID);
    }

    // 5. SE layer 2 on x2 [.,.,512]
    colmean_kernel<<<Bn, HID, 0, stream>>>(x2, ybar2, HID);
    se_mlp_kernel<<<Bn, HID, (HID / 4) * sizeof(float), stream>>>(ybar2, se2w1, se2w2, scale2, HID, HID / 4);
    {
        int total = (int)(M * HID);
        scale_h16_kernel<<<(total + 255) / 256, 256, 0, stream>>>(x2, scale2, x2h, HID, total);
    }

    // 6. output head: hout = x2s @ Wout, dots, attention -> d_out (ELU fused)
    {
        int tiles = (int)(M / 16) * (NOUT / 16);             // 4096
        gemm_h16_kernel<<<tiles / 4, 128, 0, stream>>>(x2h, Wout, hout, (int)M, HID, NOUT);
        edot_kernel<<<(int)(M / 256), 256, 0, stream>>>(hout, Aout, eS2, eD2, (int)M, NOUT);
        int atiles = Bn * (Nn / 16);                         // 512
        attn_kernel<NOUT><<<atiles / 4, 128, 0, stream>>>(hout, eS2, eD2, adj,
                                                          (float*)d_out, NOUT);
    }
}
